// MultiHeadSpatialAttention_84688165142780
// MI455X (gfx1250) — matrix operand order repaired and checked by reading, not compiled, not run on hardware
//
#include <hip/hip_runtime.h>

typedef __attribute__((ext_vector_type(16))) _Float16 v16h;
typedef __attribute__((ext_vector_type(8)))  _Float16 v8h;
typedef __attribute__((ext_vector_type(8)))  float    v8f;

#define B_   4
#define C_   256
#define NH   8
#define HD   32
#define L_   2304
#define LT   144            // L_/16
#define KBLK 72             // L_/32
#define QSCALE 0.17677669529663687f   // 1/sqrt(32)

// ---- WMMA helpers -------------------------------------------------------

static __device__ __forceinline__ v8f wmma_f16(v16h a, v16h b, v8f c) {
  // (neg_a, A, neg_b, B, c_mod, C, reuse_a, reuse_b)
  return __builtin_amdgcn_wmma_f32_16x16x32_f16(false, a, false, b, (short)0, c,
                                                false, false);
}

// A-operand (16x32 f16) load from a row-major [rows][rs] f16 array.
// Contiguous k order (B loads use the same): lane(0-15)=row M; elems 0..15 = K (c0 + half*16 + e).
static __device__ __forceinline__ v16h load_a_tile(const _Float16* base, int rs,
                                                   int row, int c0, int half) {
  union { v16h v; v8h h[2]; } u;
  const _Float16* p = base + row * rs + c0 + half * 16;
  u.h[0] = *(const v8h*)(p);
  u.h[1] = *(const v8h*)(p + 8);
  return u.v;
}

// 16B global -> LDS async copy (ASYNCcnt-tracked, CDNA5 TDM-lite path).
// the LDS address is taken from the vdst VGPR; low 32 bits of a generic pointer to a
// __shared__ object are exactly the LDS byte offset.
static __device__ __forceinline__ void async_copy16(const _Float16* g,
                                                    _Float16* l) {
  unsigned lds_off = (unsigned)(unsigned long long)(void*)l;
  asm volatile("global_load_async_to_lds_b128 %0, %1, off"
               :
               : "v"(lds_off), "v"((unsigned long long)g)
               : "memory");
}

static __device__ __forceinline__ void wait_asynccnt0() {
  asm volatile("s_wait_asynccnt 0x0" ::: "memory");
}

static __device__ __forceinline__ void lds_fence_wave() {
  // intra-wave LDS ordering: stop compiler reordering + drain DScnt
  asm volatile("s_wait_dscnt 0x0" ::: "memory");
}

// ---- kernel 0a: x[b][c][l] f32 -> xT[b][l][c] f16 (LDS tile transpose) --

__global__ void transpose_x_kernel(const float* __restrict__ x,
                                   _Float16* __restrict__ xT) {
  __shared__ float tile[16][17];
  int b  = blockIdx.z;
  int c0 = blockIdx.y * 16;
  int l0 = blockIdx.x * 16;
  int tx = threadIdx.x & 15, ty = threadIdx.x >> 4;
  tile[ty][tx] = x[(b * C_ + c0 + ty) * L_ + l0 + tx];
  __syncthreads();
  xT[(b * L_ + l0 + ty) * C_ + c0 + tx] = (_Float16)tile[tx][ty];
}

// ---- kernel 0b: convert 4 weight matrices (f32 -> f16, row-major) -------

__global__ void convert_w_kernel(const float* __restrict__ wq,
                                 const float* __restrict__ wk,
                                 const float* __restrict__ wv,
                                 const float* __restrict__ wo,
                                 _Float16* __restrict__ wf16) {
  int t = blockIdx.x * 256 + threadIdx.x;      // 4*65536 total
  int m = t >> 16, i = t & 65535;
  const float* src = (m == 0) ? wq : ((m == 1) ? wk : ((m == 2) ? wv : wo));
  wf16[t] = (_Float16)src[i];
}

// ---- kernel A: fused QKV projection (WMMA GEMM + bias + relayout) -------
// Q: [b][h][l][d] f16 (pre-scaled by 1/sqrt(d)),  K: [b][h][l][d] f16,
// V: [b][h][d][L] f16

__global__ void qkv_proj_kernel(const _Float16* __restrict__ xT,
                                const _Float16* __restrict__ wf16,
                                const float* __restrict__ bq,
                                const float* __restrict__ bk,
                                const float* __restrict__ bv,
                                _Float16* __restrict__ Q,
                                _Float16* __restrict__ K,
                                _Float16* __restrict__ V) {
  int lid = threadIdx.x & 31, wid = threadIdx.x >> 5;
  int gid = blockIdx.x * 4 + wid;           // 3*4*16*144 waves
  int lt  = gid % LT;  int tmp = gid / LT;
  int ot  = tmp & 15;  tmp >>= 4;
  int b   = tmp & 3;   int proj = tmp >> 2; // 0=q 1=k 2=v
  int half = lid >> 4, col = lid & 15;
  int o0 = ot * 16, l0 = lt * 16;

  const _Float16* Wm   = wf16 + proj * 65536;
  const float*    bias = (proj == 0) ? bq : ((proj == 1) ? bk : bv);
  const _Float16* xb   = xT + b * (L_ * C_);

  v8f acc = {};
  for (int kk = 0; kk < 8; ++kk) {
    v16h a  = load_a_tile(Wm, C_, o0 + col, kk * 32, half);
    v16h bm = *(const v16h*)(xb + (l0 + col) * C_ + kk * 32 + half * 16);
    acc = wmma_f16(a, bm, acc);
  }

  float bvv[8];
  #pragma unroll
  for (int r = 0; r < 8; ++r) bvv[r] = bias[o0 + half * 8 + r];

  int h = o0 >> 5;                 // tile never crosses a head (16 | 32)
  if (proj == 2) {
    // V layout [b][h][d][L]: per-element f16 stores (stride L)
    #pragma unroll
    for (int r = 0; r < 8; ++r) {
      int d = (o0 & 31) + half * 8 + r;
      V[((b * NH + h) * HD + d) * L_ + (l0 + col)] = (_Float16)(acc[r] + bvv[r]);
    }
  } else {
    float sc = (proj == 0) ? QSCALE : 1.0f;
    v8h pk;
    #pragma unroll
    for (int r = 0; r < 8; ++r) pk[r] = (_Float16)((acc[r] + bvv[r]) * sc);
    _Float16* dst = (proj == 0 ? Q : K) +
                    ((b * NH + h) * L_ + (l0 + col)) * HD + (o0 & 31) + half * 8;
    *(v8h*)dst = pk;               // 16B packed store (d contiguous)
  }
}

// ---- kernel B: fused flash attention ------------------------------------
// One block = one (b,h) and 4 consecutive 16-query strips (one per wave).
// K/V tiles (32 keys) are staged into LDS once per block via async loads,
// double-buffered; all 4 waves consume them from LDS.
// att out: [b][l][c] f16 with c = h*32+d.

__global__ void attn_kernel(const _Float16* __restrict__ Q,
                            const _Float16* __restrict__ K,
                            const _Float16* __restrict__ V,
                            _Float16* __restrict__ att) {
  __shared__ __align__(64) _Float16 Kt[2][32 * 32];  // [buf][key][d], 2KB each
  __shared__ __align__(64) _Float16 Vt[2][32 * 32];  // [buf][d][key], 2KB each
  __shared__ __align__(64) _Float16 Pt[4][16 * 32];  // per-wave P tile, 1KB

  int lid = threadIdx.x & 31, wid = threadIdx.x >> 5;
  int bid = blockIdx.x;                     // 4*8*36 blocks
  int qg  = bid % (LT / 4);  int tmp = bid / (LT / 4);
  int h   = tmp & 7;         int b = tmp >> 3;
  int half = lid >> 4, col = lid & 15;
  int q0 = (qg * 4 + wid) * 16;

  const _Float16* Qb = Q + (b * NH + h) * L_ * HD;
  const _Float16* Kb = K + (b * NH + h) * L_ * HD;
  const _Float16* Vb = V + (b * NH + h) * HD * L_;
  _Float16* P = Pt[wid];

  // cooperative async tile loaders (wave 0: K, wave 1: V)
  auto loadK = [&](int k0, int bf) {
    // 32 keys x 64B, fully contiguous in global: 4 x (32 lanes x 16B)
    #pragma unroll
    for (int i = 0; i < 4; ++i)
      async_copy16(Kb + k0 * HD + i * 256 + lid * 8,
                   &Kt[bf][i * 256 + lid * 8]);
  };
  auto loadV = [&](int k0, int bf) {
    // 32 d-rows x 64B, row stride L_ in global: lane -> (d, 16B segment)
    int d = lid >> 2, seg = lid & 3;
    #pragma unroll
    for (int i = 0; i < 4; ++i)
      async_copy16(Vb + (i * 8 + d) * L_ + k0 + seg * 8,
                   &Vt[bf][(i * 8 + d) * 32 + seg * 8]);
  };

  v16h aq = load_a_tile(Qb, HD, q0 + col, 0, half);   // 16 queries x d=32
  v8f acc0 = {}, acc1 = {};
  float m[8], s[8];
  #pragma unroll
  for (int r = 0; r < 8; ++r) { m[r] = -3.0e38f; s[r] = 0.0f; }

  // prologue: stage tile 0
  if (wid == 0) loadK(0, 0);
  else if (wid == 1) loadV(0, 0);
  if (wid < 2) wait_asynccnt0();
  __syncthreads();

  for (int kb = 0; kb < KBLK; ++kb) {
    int cur = kb & 1, nxt = cur ^ 1;
    // prefetch next tile while computing on current
    if (kb + 1 < KBLK) {
      if (wid == 0) loadK((kb + 1) * 32, nxt);
      else if (wid == 1) loadV((kb + 1) * 32, nxt);
    }
    const _Float16* Kc = Kt[cur];
    const _Float16* Vc = Vt[cur];

    // scores: two 16x16 tiles, K=32 (=head_dim) per WMMA, B-operand from LDS
    v16h bk0 = *(const v16h*)(Kc + col * 32 + half * 16);
    v16h bk1 = *(const v16h*)(Kc + (16 + col) * 32 + half * 16);
    v8f z = {};
    v8f s0 = wmma_f16(aq, bk0, z);
    v8f s1 = wmma_f16(aq, bk1, z);

    float p0[8], p1[8];
    #pragma unroll
    for (int r = 0; r < 8; ++r) {
      // row = r + 8*half lives across the 16 lanes of this half-wave
      float t = fmaxf(s0[r], s1[r]);
      t = fmaxf(t, __shfl_xor(t, 1));
      t = fmaxf(t, __shfl_xor(t, 2));
      t = fmaxf(t, __shfl_xor(t, 4));
      t = fmaxf(t, __shfl_xor(t, 8));
      float mn = fmaxf(m[r], t);
      float cf = __expf(m[r] - mn);
      p0[r] = __expf(s0[r] - mn);
      p1[r] = __expf(s1[r] - mn);
      float rs = p0[r] + p1[r];
      rs += __shfl_xor(rs, 1);
      rs += __shfl_xor(rs, 2);
      rs += __shfl_xor(rs, 4);
      rs += __shfl_xor(rs, 8);
      s[r] = s[r] * cf + rs;
      m[r] = mn;
      acc0[r] *= cf;
      acc1[r] *= cf;
    }

    // C-layout (row across lanes) -> A-layout (row within lane) via
    // per-wave LDS tile; intra-wave only, so no block barrier needed.
    #pragma unroll
    for (int r = 0; r < 8; ++r) {
      P[(r + half * 8) * 32 + col]      = (_Float16)p0[r];
      P[(r + half * 8) * 32 + 16 + col] = (_Float16)p1[r];
    }
    lds_fence_wave();
    v16h ap = load_a_tile(P, 32, col, 0, half);       // P: 16q x 32k
    lds_fence_wave();

    // P (16x32) x V (32 keys x 16 d) for both d halves, B-operand from LDS
    v16h bv0 = *(const v16h*)(Vc + col * 32 + half * 16);
    v16h bv1 = *(const v16h*)(Vc + (16 + col) * 32 + half * 16);
    acc0 = wmma_f16(ap, bv0, acc0);
    acc1 = wmma_f16(ap, bv1, acc1);

    // make sure next buffer landed, then release current buffer
    if (kb + 1 < KBLK && wid < 2) wait_asynccnt0();
    __syncthreads();
  }

  #pragma unroll
  for (int r = 0; r < 8; ++r) {
    float inv = 1.0f / s[r];
    int q = q0 + r + half * 8;
    _Float16* dst = att + (b * L_ + q) * C_ + h * HD;
    dst[col]      = (_Float16)(acc0[r] * inv);
    dst[16 + col] = (_Float16)(acc1[r] * inv);
  }
}

// ---- kernel C: output projection (WMMA GEMM + bias, fp32 out) -----------

__global__ void out_proj_kernel(const _Float16* __restrict__ att,
                                const _Float16* __restrict__ woh,
                                const float* __restrict__ bo,
                                float* __restrict__ out) {
  int lid = threadIdx.x & 31, wid = threadIdx.x >> 5;
  int gid = blockIdx.x * 4 + wid;           // 4*16*144 waves
  int lt  = gid % LT;  int tmp = gid / LT;
  int ot  = tmp & 15;  int b = tmp >> 4;
  int half = lid >> 4, col = lid & 15;
  int o0 = ot * 16, l0 = lt * 16;

  const _Float16* ab = att + b * (L_ * C_);
  v8f acc = {};
  for (int kk = 0; kk < 8; ++kk) {
    v16h a  = load_a_tile(woh, C_, o0 + col, kk * 32, half);
    v16h bm = *(const v16h*)(ab + (l0 + col) * C_ + kk * 32 + half * 16);
    acc = wmma_f16(a, bm, acc);
  }
  #pragma unroll
  for (int r = 0; r < 8; ++r) {
    int o = o0 + r + half * 8;
    out[(b * C_ + o) * L_ + l0 + col] = acc[r] + bo[o];
  }
}

// ---- launch --------------------------------------------------------------

extern "C" void kernel_launch(void* const* d_in, const int* in_sizes, int n_in,
                              void* d_out, int out_size, void* d_ws, size_t ws_size,
                              hipStream_t stream) {
  const float* x  = (const float*)d_in[0];
  const float* wq = (const float*)d_in[1];
  const float* bq = (const float*)d_in[2];
  const float* wk = (const float*)d_in[3];
  const float* bk = (const float*)d_in[4];
  const float* wv = (const float*)d_in[5];
  const float* bv = (const float*)d_in[6];
  const float* wo = (const float*)d_in[7];
  const float* bo = (const float*)d_in[8];
  float* out = (float*)d_out;

  char* ws = (char*)d_ws;
  const size_t SZ_XT  = (size_t)B_ * L_ * C_ * sizeof(_Float16);      // 4.5 MiB
  const size_t SZ_W   = (size_t)4 * C_ * C_ * sizeof(_Float16);       // 512 KiB
  const size_t SZ_QKV = (size_t)B_ * NH * L_ * HD * sizeof(_Float16); // 4.5 MiB
  _Float16* xT   = (_Float16*)(ws);
  _Float16* wf16 = (_Float16*)(ws + SZ_XT);
  _Float16* Q    = (_Float16*)(ws + SZ_XT + SZ_W);
  _Float16* K    = (_Float16*)(ws + SZ_XT + SZ_W + SZ_QKV);
  _Float16* V    = (_Float16*)(ws + SZ_XT + SZ_W + 2 * SZ_QKV);
  _Float16* att  = (_Float16*)(ws + SZ_XT + SZ_W + 3 * SZ_QKV);
  _Float16* woh  = wf16 + 3 * C_ * C_;

  transpose_x_kernel<<<dim3(LT, C_ / 16, B_), 256, 0, stream>>>(x, xT);
  convert_w_kernel<<<(4 * C_ * C_) / 256, 256, 0, stream>>>(wq, wk, wv, wo, wf16);

  // 3 proj * 4 batch * 16 o-tiles * 144 l-tiles waves, 4 waves/block
  qkv_proj_kernel<<<(3 * B_ * 16 * LT) / 4, 128, 0, stream>>>(
      xT, wf16, bq, bk, bv, Q, K, V);

  // one block per (b,h,4 q-tiles): 4*8*36 blocks
  attn_kernel<<<B_ * NH * (LT / 4), 128, 0, stream>>>(Q, K, V, att);

  // 4 * 16 * 144 waves, 4 waves/block
  out_proj_kernel<<<(B_ * 16 * LT) / 4, 128, 0, stream>>>(att, woh, bo, out);
}
